// MultiHeadAttention_44616120270963
// MI455X (gfx1250) — matrix operand order repaired and checked by reading, not compiled, not run on hardware
//
#include <hip/hip_runtime.h>
#include <hip/hip_bf16.h>
#include <math.h>

typedef __attribute__((ext_vector_type(16))) __bf16 v16bf;
typedef __attribute__((ext_vector_type(8)))  float  v8f;

union Frag {
    v16bf v;
    uint4 q[2];
    unsigned short u[16];
};

__device__ __forceinline__ unsigned short f2bf(float f) {
    unsigned int u = __builtin_bit_cast(unsigned int, f);
    u += 0x7FFFu + ((u >> 16) & 1u);   // round-to-nearest-even
    return (unsigned short)(u >> 16);
}
__device__ __forceinline__ float bf2f(unsigned short s) {
    unsigned int u = ((unsigned int)s) << 16;
    return __builtin_bit_cast(float, u);
}

static constexpr int S   = 2048;  // sequence
static constexpr int DIM = 2048;  // model dim
static constexpr int HD  = 64;    // head dim
static constexpr int KVD = 512;   // n_kv_heads * head_dim

// ---------------------------------------------------------------------------
// fp32 -> bf16 convert
__global__ void cvt_kernel(const float* __restrict__ src,
                           unsigned short* __restrict__ dst, int n) {
    int i = blockIdx.x * blockDim.x + threadIdx.x;
    if (i < n) dst[i] = f2bf(src[i]);
}

// fp32 [K x N] -> bf16 transposed [N x K]
__global__ void tcvt_kernel(const float* __restrict__ src,
                            unsigned short* __restrict__ dst, int K, int N) {
    int i = blockIdx.x * blockDim.x + threadIdx.x;
    if (i >= N * K) return;
    int n = i / K, k = i % K;
    dst[i] = f2bf(src[(size_t)k * N + n]);
}

// bf16 [rows x cols] -> bf16 [cols x rows]
__global__ void tr_kernel(const unsigned short* __restrict__ src,
                          unsigned short* __restrict__ dst, int rows, int cols) {
    int i = blockIdx.x * blockDim.x + threadIdx.x;
    if (i >= rows * cols) return;
    int c = i / rows, s = i % rows;
    dst[i] = src[(size_t)s * cols + c];
}

// ---------------------------------------------------------------------------
// RoPE (interleaved rotate_half, cos/sin generated on the fly), bf16->bf16
__global__ void rope_kernel(const unsigned short* __restrict__ src,
                            unsigned short* __restrict__ dst, int ncols) {
    int idx = blockIdx.x * blockDim.x + threadIdx.x;
    if (idx >= S * ncols) return;
    int s = idx / ncols, c = idx % ncols;
    int d  = c & (HD - 1);
    int hb = c - d;          // head base column
    int i  = d & 31;
    float inv = __powf(10000.0f, -(float)(2 * i) * (1.0f / 64.0f));
    float ang = (float)s * inv;
    float cs = __cosf(ang), sn = __sinf(ang);
    float x = bf2f(src[idx]);
    float pr = (d < 32) ? -bf2f(src[(size_t)s * ncols + hb + 2 * d + 1])
                        :  bf2f(src[(size_t)s * ncols + hb + 2 * (d - 32)]);
    dst[idx] = f2bf(x * cs + pr * sn);
}

// ---------------------------------------------------------------------------
// bf16 WMMA GEMM: C[MxN] = A[MxK] * B[KxN], B given transposed as BT[NxK].
// Block = 256 threads = 8 waves (2 in M x 4 in N); block tile 128x256;
// wave tile 64x64 = 4x4 subtiles of 16x16; K stepped by 32.
// A fragments (4) are loaded once per K-step and reused across the 4
// N-subtiles; B fragments are loaded one at a time to bound live VGPRs.
__global__ __launch_bounds__(256)
void gemm_bf16_kernel(const unsigned short* __restrict__ A,
                      const unsigned short* __restrict__ BT,
                      void* __restrict__ Cout,
                      int M, int N, int K, int out_f32) {
    const int tid  = threadIdx.x;
    const int wave = tid >> 5, lane = tid & 31;
    const int lo = lane & 15, hi = lane >> 4;
    const int bm0 = blockIdx.y * 128, bn0 = blockIdx.x * 256;
    const int wm = (wave >> 2) * 64, wn = (wave & 3) * 64;

    v8f acc[4][4];
#pragma unroll
    for (int i = 0; i < 4; i++)
#pragma unroll
        for (int j = 0; j < 4; j++) {
            v8f z = {0.f, 0.f, 0.f, 0.f, 0.f, 0.f, 0.f, 0.f};
            acc[i][j] = z;
        }

    for (int k0 = 0; k0 < K; k0 += 32) {
        Frag a[4];
#pragma unroll
        for (int ms = 0; ms < 4; ms++) {
            const unsigned short* p =
                A + (size_t)(bm0 + wm + ms * 16 + lo) * K + k0 + hi * 8;
            a[ms].q[0] = *(const uint4*)p;
            a[ms].q[1] = *(const uint4*)(p + 16);
            __builtin_prefetch(p + 32, 0, 1);          // speculative, branchless
        }
#pragma unroll
        for (int ns = 0; ns < 4; ns++) {
            Frag b;
            const unsigned short* p =
                BT + (size_t)(bn0 + wn + ns * 16 + lo) * K + k0 + hi * 8;
            b.q[0] = *(const uint4*)p;
            b.q[1] = *(const uint4*)(p + 16);
            __builtin_prefetch(p + 32, 0, 1);
#pragma unroll
            for (int ms = 0; ms < 4; ms++)
                acc[ms][ns] = __builtin_amdgcn_wmma_f32_16x16x32_bf16(
                    false, a[ms].v, false, b.v,
                    (short)0, acc[ms][ns], false, false);
        }
    }

#pragma unroll
    for (int ms = 0; ms < 4; ms++)
#pragma unroll
        for (int ns = 0; ns < 4; ns++)
#pragma unroll
            for (int r = 0; r < 8; r++) {
                size_t row = bm0 + wm + ms * 16 + r + 8 * hi;
                size_t col = bn0 + wn + ns * 16 + lo;
                float v = acc[ms][ns][r];
                if (out_f32) ((float*)Cout)[row * N + col] = v;
                else         ((unsigned short*)Cout)[row * N + col] = f2bf(v);
            }
}

// ---------------------------------------------------------------------------
// Flash attention (causal, GQA 4:1). One wave owns 16 query rows; 32-key
// blocks. Scores: 2 WMMAs per 16x16 tile (K=64 split 2x32). Softmax stats via
// shfl_xor over the 16-lane half-groups. P restaged through per-wave LDS to
// become an A-fragment for the PxV WMMAs (V pre-transposed to [d][s]).
__global__ __launch_bounds__(256)
void flash_kernel(const unsigned short* __restrict__ Q,
                  const unsigned short* __restrict__ Kr,
                  const unsigned short* __restrict__ Vt,
                  unsigned short* __restrict__ O) {
    __shared__ alignas(16) unsigned short pbuf[8][16][32];

    const int tid  = threadIdx.x;
    const int wave = tid >> 5, lane = tid & 31;
    const int lo = lane & 15, hi = lane >> 4;
    const int head  = blockIdx.y;
    const int kvb   = (head >> 2) * HD;              // kv-head column base
    const int qbase = blockIdx.x * 128 + wave * 16;

    Frag aq[2];
    {
        const unsigned short* p = Q + (size_t)(qbase + lo) * DIM + head * HD;
        aq[0].q[0] = *(const uint4*)(p + hi * 8);
        aq[0].q[1] = *(const uint4*)(p + 16 + hi * 8);
        aq[1].q[0] = *(const uint4*)(p + 32 + hi * 8);
        aq[1].q[1] = *(const uint4*)(p + 48 + hi * 8);
    }

    v8f o[4];
#pragma unroll
    for (int d4 = 0; d4 < 4; d4++) {
        v8f z = {0.f, 0.f, 0.f, 0.f, 0.f, 0.f, 0.f, 0.f};
        o[d4] = z;
    }
    float mrow[8], lrow[8];
#pragma unroll
    for (int r = 0; r < 8; r++) { mrow[r] = -__builtin_inff(); lrow[r] = 0.f; }

    const int nkb = (qbase + 15) / 32 + 1;
    for (int kb = 0; kb < nkb; kb++) {
        const int kidx0 = kb * 32;

        v8f sc[2];
#pragma unroll
        for (int t = 0; t < 2; t++) {
            Frag bk[2];
#pragma unroll
            for (int f = 0; f < 2; f++) {
                const unsigned short* p =
                    Kr + (size_t)(kidx0 + t * 16 + lo) * KVD + kvb + f * 32 + hi * 8;
                bk[f].q[0] = *(const uint4*)p;
                bk[f].q[1] = *(const uint4*)(p + 16);
            }
            v8f z = {0.f, 0.f, 0.f, 0.f, 0.f, 0.f, 0.f, 0.f};
            z = __builtin_amdgcn_wmma_f32_16x16x32_bf16(
                false, aq[0].v, false, bk[0].v, (short)0, z, false, false);
            z = __builtin_amdgcn_wmma_f32_16x16x32_bf16(
                false, aq[1].v, false, bk[1].v, (short)0, z, false, false);
            sc[t] = z;
        }

        // scale + causal mask
#pragma unroll
        for (int t = 0; t < 2; t++)
#pragma unroll
            for (int r = 0; r < 8; r++) {
                int row = qbase + r + 8 * hi;
                int col = kidx0 + t * 16 + lo;
                float v = sc[t][r] * 0.125f;              // 1/sqrt(64)
                sc[t][r] = (col <= row) ? v : -__builtin_inff();
            }

        // online softmax
#pragma unroll
        for (int r = 0; r < 8; r++) {
            float bm = fmaxf(sc[0][r], sc[1][r]);
            bm = fmaxf(bm, __shfl_xor(bm, 1, 32));
            bm = fmaxf(bm, __shfl_xor(bm, 2, 32));
            bm = fmaxf(bm, __shfl_xor(bm, 4, 32));
            bm = fmaxf(bm, __shfl_xor(bm, 8, 32));
            float mn = fmaxf(mrow[r], bm);
            float scale = __expf(mrow[r] - mn);
            float p0 = __expf(sc[0][r] - mn);
            float p1 = __expf(sc[1][r] - mn);
            float rs = p0 + p1;
            rs += __shfl_xor(rs, 1, 32);
            rs += __shfl_xor(rs, 2, 32);
            rs += __shfl_xor(rs, 4, 32);
            rs += __shfl_xor(rs, 8, 32);
            lrow[r] = lrow[r] * scale + rs;
            mrow[r] = mn;
#pragma unroll
            for (int d4 = 0; d4 < 4; d4++) o[d4][r] *= scale;
            pbuf[wave][r + 8 * hi][lo]      = f2bf(p0);
            pbuf[wave][r + 8 * hi][16 + lo] = f2bf(p1);
        }
#if __has_builtin(__builtin_amdgcn_wave_barrier)
        __builtin_amdgcn_wave_barrier();   // per-wave LDS RAW; DS is in-order
#endif
        Frag ap;
        ap.q[0] = *(const uint4*)&pbuf[wave][lo][hi * 8];
        ap.q[1] = *(const uint4*)&pbuf[wave][lo][16 + hi * 8];

#pragma unroll
        for (int d4 = 0; d4 < 4; d4++) {
            Frag bv;
            const unsigned short* p =
                Vt + (size_t)(kvb + d4 * 16 + lo) * S + kidx0 + hi * 8;
            bv.q[0] = *(const uint4*)p;
            bv.q[1] = *(const uint4*)(p + 16);
            o[d4] = __builtin_amdgcn_wmma_f32_16x16x32_bf16(
                false, ap.v, false, bv.v, (short)0, o[d4], false, false);
        }
    }

#pragma unroll
    for (int d4 = 0; d4 < 4; d4++)
#pragma unroll
        for (int r = 0; r < 8; r++) {
            size_t row = qbase + r + 8 * hi;
            float val = o[d4][r] / lrow[r];
            O[row * DIM + head * HD + d4 * 16 + lo] = f2bf(val);
        }
}

// ---------------------------------------------------------------------------
extern "C" void kernel_launch(void* const* d_in, const int* in_sizes, int n_in,
                              void* d_out, int out_size, void* d_ws, size_t ws_size,
                              hipStream_t stream) {
    const float* x  = (const float*)d_in[0];
    // d_in[1] = mask (causal; recomputed in-kernel)
    const float* wq = (const float*)d_in[2];
    const float* wk = (const float*)d_in[3];
    const float* wv = (const float*)d_in[4];
    const float* wo = (const float*)d_in[5];
    float* out = (float*)d_out;

    char* ws = (char*)d_ws;
    size_t off = 0;
    auto alloc = [&](size_t bytes) -> unsigned short* {
        unsigned short* p = (unsigned short*)(ws + off);
        off += (bytes + 255) & ~(size_t)255;
        return p;
    };
    unsigned short* xb  = alloc((size_t)S * DIM * 2);
    unsigned short* wqT = alloc((size_t)DIM * DIM * 2);
    unsigned short* wkT = alloc((size_t)KVD * DIM * 2);
    unsigned short* wvT = alloc((size_t)KVD * DIM * 2);
    unsigned short* woT = alloc((size_t)DIM * DIM * 2);
    unsigned short* qb  = alloc((size_t)S * DIM * 2);
    unsigned short* kbf = alloc((size_t)S * KVD * 2);
    unsigned short* vbf = alloc((size_t)S * KVD * 2);
    unsigned short* qr  = alloc((size_t)S * DIM * 2);
    unsigned short* kr  = alloc((size_t)S * KVD * 2);
    unsigned short* vt  = alloc((size_t)KVD * S * 2);
    unsigned short* ob  = alloc((size_t)S * DIM * 2);

    const int T = 256;
    cvt_kernel <<<(S * DIM + T - 1) / T, T, 0, stream>>>(x, xb, S * DIM);
    tcvt_kernel<<<(DIM * DIM + T - 1) / T, T, 0, stream>>>(wq, wqT, DIM, DIM);
    tcvt_kernel<<<(KVD * DIM + T - 1) / T, T, 0, stream>>>(wk, wkT, DIM, KVD);
    tcvt_kernel<<<(KVD * DIM + T - 1) / T, T, 0, stream>>>(wv, wvT, DIM, KVD);
    tcvt_kernel<<<(DIM * DIM + T - 1) / T, T, 0, stream>>>(wo, woT, DIM, DIM);

    gemm_bf16_kernel<<<dim3(DIM / 256, S / 128), T, 0, stream>>>(
        xb, wqT, qb, S, DIM, DIM, 0);
    gemm_bf16_kernel<<<dim3(KVD / 256, S / 128), T, 0, stream>>>(
        xb, wkT, kbf, S, KVD, DIM, 0);
    gemm_bf16_kernel<<<dim3(KVD / 256, S / 128), T, 0, stream>>>(
        xb, wvT, vbf, S, KVD, DIM, 0);

    rope_kernel<<<(S * DIM + T - 1) / T, T, 0, stream>>>(qb, qr, DIM);
    rope_kernel<<<(S * KVD + T - 1) / T, T, 0, stream>>>(kbf, kr, KVD);
    tr_kernel  <<<(S * KVD + T - 1) / T, T, 0, stream>>>(vbf, vt, S, KVD);

    flash_kernel<<<dim3(S / 128, 32), T, 0, stream>>>(qr, kr, vt, ob);

    gemm_bf16_kernel<<<dim3(DIM / 256, S / 128), T, 0, stream>>>(
        ob, woT, out, S, DIM, DIM, 1);
}
